// PointerDecoder_29068338659944
// MI455X (gfx1250) — matrix operand order repaired and checked by reading, not compiled, not run on hardware
//
#include <hip/hip_runtime.h>
#include <hip/hip_bf16.h>
#include <stddef.h>

// ---------------------------------------------------------------- types
typedef __attribute__((ext_vector_type(16))) __bf16        v16bf;
typedef __attribute__((ext_vector_type(8)))  float         v8f;
typedef __attribute__((ext_vector_type(8)))  unsigned short u16x8;
typedef __attribute__((ext_vector_type(16))) unsigned short u16x16;

#define Bn   128
#define Sn   1024
#define Hn   256
#define NEGV 1e9f

__device__ __forceinline__ unsigned short f32_to_bf16(float x) {
  unsigned u = __builtin_bit_cast(unsigned, x);
  unsigned r = 0x7FFFu + ((u >> 16) & 1u);          // round-to-nearest-even
  return (unsigned short)((u + r) >> 16);
}

// two f32 -> packed bf16x2 in one VALU op (gfx1250 v_cvt_pk_bf16_f32)
__device__ __forceinline__ unsigned pack_bf16x2(float lo, float hi) {
  unsigned p;
  asm("v_cvt_pk_bf16_f32 %0, %1, %2" : "=v"(p) : "v"(lo), "v"(hi));
  return p;
}

// hardware tanh: single TRANS op on CDNA5 instead of libm exp expansion
__device__ __forceinline__ float fast_tanh(float x) {
#if __has_builtin(__builtin_amdgcn_tanhf)
  return __builtin_amdgcn_tanhf(x);
#else
  float y;
  // v_nop satisfies the TRANS co-execution hazard (1 indep op before use)
  asm volatile("v_tanh_f32 %0, %1\n\tv_nop" : "=v"(y) : "v"(x));
  return y;
#endif
}

// ------------------------------------------------- 1) weight conversion
__global__ __launch_bounds__(256)
void convert_weights(const float* __restrict__ wg, const float* __restrict__ wp,
                     unsigned short* __restrict__ wg16,
                     unsigned short* __restrict__ wp16) {
  int i = blockIdx.x * blockDim.x + threadIdx.x;
  if (i < Hn * Hn) {
    wg16[i] = f32_to_bf16(wg[i]);
    wp16[i] = f32_to_bf16(wp[i]);
  }
}

// ------------------------------------------- 2) q = x @ W^T  (tiny matvec)
__global__ __launch_bounds__(256)
void matvec_qT(const float* __restrict__ x,   // [B,H]
               const float* __restrict__ w,   // [H,H]; out[b,h]=sum_d x[b,d]*w[h,d]
               float* __restrict__ out) {     // [B,H]
  __shared__ float xs[Hn];
  int b = blockIdx.x, h = threadIdx.x;
  xs[h] = x[b * Hn + h];
  __syncthreads();
  const float* wr = w + (size_t)h * Hn;
  float acc = 0.f;
  #pragma unroll 8
  for (int d = 0; d < Hn; ++d) acc = fmaf(xs[d], wr[d], acc);
  out[b * Hn + h] = acc;
}

// ---- 3/6) fused:  scores[b,s] = v . tanh(ref@W^T + bias + q[b]) - mask*NEG
//  block = 512 threads (16 wave32), grid = (S/16, B)
//  each wave owns one 16-wide N-tile; K-loop of 8 x wmma_f32_16x16x32_bf16
__global__ __launch_bounds__(512)
void scores_wmma(const float*          __restrict__ ref,   // [B,S,H] f32
                 const unsigned short* __restrict__ w16,   // [H,H] bf16, w[h][d]
                 const float*          __restrict__ bias,  // [H]
                 const float*          __restrict__ qvec,  // [B,H]
                 const float*          __restrict__ vvec,  // [H]
                 const int*            __restrict__ mask,  // [B,S]
                 float*                __restrict__ out) { // [B,S]
  constexpr int LDSK = Hn + 8;                         // padded bf16 row stride
  __shared__ __attribute__((aligned(16))) unsigned short Atile[16 * LDSK];
  __shared__ float sscore[16];

  const int s0   = blockIdx.x * 16;
  const int b    = blockIdx.y;
  const int tid  = threadIdx.x;
  const int wave = tid >> 5;
  const int lane = tid & 31;

  if (tid < 16) sscore[tid] = 0.f;

  // ---- stage A tile: 16 rows x 256 cols f32 -> bf16 in LDS
  //      (coalesced global b128 loads, packed cvt, merged ds_store)
  {
    const int r = tid >> 5;                 // row 0..15 (one wave per row)
    const int c = (tid & 31) * 8;           // 8 consecutive elems per thread
    const float* src = ref + ((size_t)b * Sn + s0 + r) * Hn + c;
    const float4* src4 = (const float4*)src;
    float4 p0 = src4[0], p1 = src4[1];
    // prefetch the tile 16 rows ahead (stream hint into the caches)
    __builtin_prefetch(src + 16 * Hn, 0, 0);
    unsigned* dst32 = (unsigned*)&Atile[r * LDSK + c];
    dst32[0] = pack_bf16x2(p0.x, p0.y);
    dst32[1] = pack_bf16x2(p0.z, p0.w);
    dst32[2] = pack_bf16x2(p1.x, p1.y);
    dst32[3] = pack_bf16x2(p1.z, p1.w);
  }
  __syncthreads();

  // ---- WMMA GEMM: this wave computes enc[s0..s0+15, n0..n0+15]
  const int n0    = wave * 16;
  const int ncol  = n0 + (lane & 15);       // C/D + B column owned by lane
  const int khalf = lane >> 4;              // 0 or 1
  const int arow  = lane & 15;              // A: M = lane % 16
  const int ac0   = khalf * 16;             // A: K sub-offset per lane half

  const unsigned short* wrow = w16 + (size_t)ncol * Hn + khalf * 16;

  v8f acc = {};
  #pragma unroll
  for (int kk = 0; kk < Hn; kk += 32) {
    union { u16x8 h[2]; v16bf v; } fa;      // A fragment: 2 x ds_load_b128
    fa.h[0] = *(const u16x8*)&Atile[arow * LDSK + kk + ac0];
    fa.h[1] = *(const u16x8*)&Atile[arow * LDSK + kk + 8 + ac0];
    union { u16x16 u; v16bf v; } fb;        // B fragment: 16 contiguous K
    fb.u = *(const u16x16*)(wrow + kk);
    acc = __builtin_amdgcn_wmma_f32_16x16x32_bf16(
        /*neg_a=*/false, fa.v, /*neg_b=*/false, fb.v,
        /*c_mod=*/(short)0, acc, /*reuse_a=*/false, /*reuse_b=*/false);
  }

  // ---- epilogue: tanh(enc + bias + q) * v, reduce over n
  const float bq = bias[ncol] + qvec[b * Hn + ncol];
  const float vn = vvec[ncol];
  #pragma unroll
  for (int r = 0; r < 8; ++r) {
    float t = fast_tanh(acc[r] + bq) * vn;
    #pragma unroll
    for (int off = 1; off < 16; off <<= 1)  // reduce 16 lanes of each half
      t += __shfl_xor(t, off, 32);
    if ((lane & 15) == 0) {
      const int m = r + 8 * khalf;          // C layout: M = r + 8*(lane>>4)
      atomicAdd(&sscore[m], t);             // ds_add_f32 across 16 waves
    }
  }
  __syncthreads();

  if (tid < 16) {
    const float mf = (float)mask[b * Sn + s0 + tid];
    out[(size_t)b * Sn + s0 + tid] = sscore[tid] - mf * NEGV;
  }
}

// ------------- 4) softmax over S + glimpse = att @ ref + query, per batch
__global__ __launch_bounds__(256)
void softmax_glimpse(const float* __restrict__ scores,  // [B,S] (masked)
                     const float* __restrict__ ref,     // [B,S,H]
                     const float* __restrict__ query,   // [B,H]
                     float* __restrict__ glimpse) {     // [B,H]
  __shared__ float att[Sn];
  __shared__ float red[256];
  const int b = blockIdx.x, t = threadIdx.x;
  const float* sb = scores + (size_t)b * Sn;

  float lmax = -3.4e38f;
  for (int s = t; s < Sn; s += 256) { float x = sb[s]; att[s] = x; lmax = fmaxf(lmax, x); }
  red[t] = lmax; __syncthreads();
  for (int o = 128; o > 0; o >>= 1) { if (t < o) red[t] = fmaxf(red[t], red[t + o]); __syncthreads(); }
  const float m = red[0]; __syncthreads();

  float lsum = 0.f;
  for (int s = t; s < Sn; s += 256) { float e = __expf(att[s] - m); att[s] = e; lsum += e; }
  red[t] = lsum; __syncthreads();
  for (int o = 128; o > 0; o >>= 1) { if (t < o) red[t] += red[t + o]; __syncthreads(); }
  const float inv = 1.f / red[0];

  // glimpse[b, t]: coalesced column walk over ref
  const float* rb = ref + (size_t)b * Sn * Hn + t;
  float acc = 0.f;
  #pragma unroll 4
  for (int s = 0; s < Sn; ++s) acc = fmaf(att[s], rb[(size_t)s * Hn], acc);
  glimpse[b * Hn + t] = acc * inv + query[b * Hn + t];
}

// ---------------------------------------------------------------- launch
extern "C" void kernel_launch(void* const* d_in, const int* in_sizes, int n_in,
                              void* d_out, int out_size, void* d_ws, size_t ws_size,
                              hipStream_t stream) {
  const float* ref   = (const float*)d_in[0];
  const float* query = (const float*)d_in[1];
  const int*   mask  = (const int*)  d_in[2];
  const float* wg    = (const float*)d_in[3];
  const float* bg    = (const float*)d_in[4];
  const float* wqg   = (const float*)d_in[5];
  const float* vg    = (const float*)d_in[6];
  const float* wp    = (const float*)d_in[7];
  const float* bp    = (const float*)d_in[8];
  const float* wqp   = (const float*)d_in[9];
  const float* vp    = (const float*)d_in[10];
  float* out = (float*)d_out;

  // workspace layout (bytes)
  char* ws = (char*)d_ws;
  unsigned short* wg16    = (unsigned short*)(ws + 0);        // 128 KB
  unsigned short* wp16    = (unsigned short*)(ws + 131072);   // 128 KB
  float*          qg      = (float*)(ws + 262144);            // 128 KB
  float*          qp      = (float*)(ws + 393216);            // 128 KB
  float*          scoresg = (float*)(ws + 524288);            // 512 KB
  float*          glimpse = (float*)(ws + 1048576);           // 128 KB
  (void)in_sizes; (void)n_in; (void)out_size; (void)ws_size;

  convert_weights<<<(Hn * Hn) / 256, 256, 0, stream>>>(wg, wp, wg16, wp16);
  matvec_qT<<<Bn, 256, 0, stream>>>(query, wqg, qg);
  scores_wmma<<<dim3(Sn / 16, Bn), 512, 0, stream>>>(ref, wg16, bg, qg, vg, mask, scoresg);
  softmax_glimpse<<<Bn, 256, 0, stream>>>(scoresg, ref, query, glimpse);
  matvec_qT<<<Bn, 256, 0, stream>>>(glimpse, wqp, qp);
  scores_wmma<<<dim3(Sn / 16, Bn), 512, 0, stream>>>(ref, wp16, bp, qp, vp, mask, out);
}
